// TopKQueryBessKGE_24197845745915
// MI455X (gfx1250) — matrix operand order repaired and checked by reading, not compiled, not run on hardware
//
#include <hip/hip_runtime.h>

// ---------------- problem constants (match reference) ----------------
#define N_SHARD     4
#define MAX_ENT     131072
#define EMB         128
#define SHARD_BS    256
#define KTOP        10
#define NB          11                 // K+1 per-shard list size
#define BAD_SCORE   (-50000.0f)
#define SHARD_COUNT (MAX_ENT - 1024)
#define NQ          (N_SHARD * SHARD_BS)   // 1024 queries

// ---------------- tiling ----------------
#define TILE_E   64                    // entities per LDS tile
#define NCHUNK   16                    // entity chunks per shard (grid.x)
#define BLOCK_Q  256                   // queries per workgroup (8 waves x 2 qtiles x 16)
#define NLISTS   (NCHUNK * 2)          // partial top-11 lists per (shard, query)
#define NCAND    (NLISTS * NB)         // 352 candidates per (shard, query)

typedef __bf16 bf16_t;
typedef __attribute__((ext_vector_type(16))) __bf16 v16bf;
typedef __attribute__((ext_vector_type(8)))  __bf16 v8bf;
typedef __attribute__((ext_vector_type(8)))  float  v8f;

union ABu { v16bf v; v8bf h[2]; };

__device__ __forceinline__ unsigned short f32_to_bf16(float f) {
  unsigned u = __builtin_bit_cast(unsigned, f);
  u += 0x7FFFu + ((u >> 16) & 1u);     // round-to-nearest-even
  return (unsigned short)(u >> 16);
}

// Register-resident top-N with replace-min; no dynamic VGPR indexing
// (predicated writes keep arrays in registers in the hot kernel).
template <int N>
__device__ __forceinline__ void tk_insert(float (&val)[N], int (&idx)[N],
                                          float &vmin, int &pmin,
                                          float s, int id) {
  if (s <= vmin) return;
#pragma unroll
  for (int j = 0; j < N; ++j)
    if (j == pmin) { val[j] = s; idx[j] = id; }
  float m = val[0]; int p = 0;
#pragma unroll
  for (int j = 1; j < N; ++j)
    if (val[j] < m) { m = val[j]; p = j; }
  vmin = m; pmin = p;
}

// ---------------- kernel 1: hr = head_emb * rel_emb, f32 -> bf16 ----------------
__global__ void hr_kernel(const int* __restrict__ relation,
                          const int* __restrict__ head,
                          const float* __restrict__ ent,
                          const float* __restrict__ rel,
                          unsigned short* __restrict__ hr) {
  const int q = blockIdx.x;            // 0..1023
  const int e = threadIdx.x;           // 0..127
  const int s = q >> 8;
  const int r = relation[q];
  const int h = head[q];
  const float v = ent[((size_t)s * MAX_ENT + h) * EMB + e] * rel[(size_t)r * EMB + e];
  hr[q * EMB + e] = f32_to_bf16(v);
}

// stage one 64x128 entity tile: global f32 -> bf16 -> LDS (coalesced b128 loads)
__device__ __forceinline__ void stage_tile(const float* __restrict__ g,
                                           unsigned short* __restrict__ lds, int tid) {
  const int base = tid * 32;           // 32 elements per thread, contiguous
#pragma unroll
  for (int j = 0; j < 8; ++j) {
    const float4 f = *(const float4*)(g + base + j * 4);
    uint2 p;
    p.x = (unsigned)f32_to_bf16(f.x) | ((unsigned)f32_to_bf16(f.y) << 16);
    p.y = (unsigned)f32_to_bf16(f.z) | ((unsigned)f32_to_bf16(f.w) << 16);
    *(uint2*)(lds + base + j * 4) = p;
  }
}

// ---------------- kernel 2: WMMA scoring + running per-query top-11 ----------------
__launch_bounds__(256, 1)
__global__ void score_topk_kernel(const unsigned short* __restrict__ hr,
                                  const float* __restrict__ ent,
                                  float* __restrict__ cand_val,
                                  int*   __restrict__ cand_idx) {
  __shared__ unsigned short ldsE[2][TILE_E * EMB];   // 2 x 16 KB bf16 entity tiles
  __shared__ float ldsC[8][16 * 16];                 // per-wave C scratch (8 KB)

  const int chunk = blockIdx.x;        // entity chunk 0..15
  const int qblk  = blockIdx.y;        // query block 0..3
  const int sneg  = blockIdx.z;        // negative shard 0..3
  const int tid   = threadIdx.x;
  const int wave  = tid >> 5;
  const int lane  = tid & 31;
  const int half  = lane >> 4;         // half-lane group (K split for A/B)
  const int l16   = lane & 15;

  const float* entS   = ent + (size_t)sneg * MAX_ENT * EMB;
  const int chunkEnt  = MAX_ENT / NCHUNK;            // 8192
  const int eBase     = chunk * chunkEnt;
  const int nTiles    = chunkEnt / TILE_E;           // 128

  // A tiles (hr), 16-bit A 16x32 layout: per lane row M=l16,
  // K = {half*16 .. +15} contiguous within each K-chunk of 32 (matches B).
  v16bf A[2][4];
#pragma unroll
  for (int qt = 0; qt < 2; ++qt) {
    const int q = qblk * BLOCK_Q + wave * 32 + qt * 16 + l16;
    const unsigned short* hp = hr + q * EMB;
#pragma unroll
    for (int kc = 0; kc < 4; ++kc) {
      A[qt][kc] = *(const v16bf*)(hp + kc * 32 + half * 16);
    }
  }

  // running top-11 lists: (qtile, half-lane entity subset) per lane
  float tv[2][NB]; int ti[2][NB];
  float tmin[2]; int tpos[2];
#pragma unroll
  for (int qt = 0; qt < 2; ++qt) {
#pragma unroll
    for (int j = 0; j < NB; ++j) { tv[qt][j] = BAD_SCORE; ti[qt][j] = MAX_ENT; }
    tmin[qt] = BAD_SCORE; tpos[qt] = 0;
  }

  stage_tile(entS + (size_t)eBase * EMB, ldsE[0], tid);
  __syncthreads();

  for (int t = 0; t < nTiles; ++t) {
    const int buf = t & 1;
    if (t + 1 < nTiles)
      stage_tile(entS + (size_t)(eBase + (t + 1) * TILE_E) * EMB, ldsE[buf ^ 1], tid);
    if (t + 2 < nTiles)
      __builtin_prefetch(entS + (size_t)(eBase + (t + 2) * TILE_E) * EMB + tid * 32, 0, 1);

#pragma unroll
    for (int et = 0; et < 4; ++et) {
      // B 32x16 bf16 layout: lane = column (entity), contiguous 16 K per half-lane.
      v16bf B[4];
      const unsigned short* bp = &ldsE[buf][(et * 16 + l16) * EMB + half * 16];
#pragma unroll
      for (int kc = 0; kc < 4; ++kc)
        B[kc] = *(const v16bf*)(bp + kc * 32);

#pragma unroll
      for (int qt = 0; qt < 2; ++qt) {
        v8f acc = {0.f, 0.f, 0.f, 0.f, 0.f, 0.f, 0.f, 0.f};
#pragma unroll
        for (int kc = 0; kc < 4; ++kc)
          acc = __builtin_amdgcn_wmma_f32_16x16x32_bf16(
              false, A[qt][kc], false, B[kc], (short)0, acc, false, false);

        // Dump C (lane holds col n=l16, rows r+8*half) to per-wave LDS scratch.
        float* cs = ldsC[wave];
#pragma unroll
        for (int r = 0; r < 8; ++r)
          cs[(r + 8 * half) * 16 + l16] = acc[r];
        asm volatile("s_wait_dscnt 0" ::: "memory");   // intra-wave LDS visibility

        // Owner-lane scan: lane owns query row l16, entities half*8..+7.
        const int ebase = eBase + t * TILE_E + et * 16 + half * 8;
#pragma unroll
        for (int j = 0; j < 8; ++j) {
          const float sc = cs[l16 * 16 + half * 8 + j];
          tk_insert<NB>(tv[qt], ti[qt], tmin[qt], tpos[qt], sc, ebase + j);
        }
        asm volatile("s_wait_dscnt 0" ::: "memory");   // reads done before next overwrite
      }
    }
    __syncthreads();
  }

  // Emit partial candidate lists.
#pragma unroll
  for (int qt = 0; qt < 2; ++qt) {
    const int q = qblk * BLOCK_Q + wave * 32 + qt * 16 + l16;
    const size_t base = ((size_t)(sneg * NQ + q) * NLISTS + (chunk * 2 + half)) * NB;
#pragma unroll
    for (int j = 0; j < NB; ++j) {
      cand_val[base + j] = tv[qt][j];
      cand_idx[base + j] = ti[qt][j];
    }
  }
}

// ---------------- kernel 3: reference-exact combine ----------------
__global__ void final_topk_kernel(const float* __restrict__ cand_val,
                                  const int* __restrict__ cand_idx,
                                  const int* __restrict__ table,
                                  int* __restrict__ out_id,
                                  float* __restrict__ out_val) {
  const int q = blockIdx.x * blockDim.x + threadIdx.x;   // global query 0..1023
  if (q >= NQ) return;

  float fv[KTOP]; int fg[KTOP];
#pragma unroll
  for (int j = 0; j < KTOP; ++j) { fv[j] = -3.0e38f; fg[j] = 0; }
  float fmin = -3.0e38f; int fpos = 0;

  for (int sneg = 0; sneg < N_SHARD; ++sneg) {
    // per-shard top-11 over 352 partial candidates (matches reference scan result)
    float tv[NB]; int ti[NB];
#pragma unroll
    for (int j = 0; j < NB; ++j) { tv[j] = BAD_SCORE; ti[j] = MAX_ENT; }
    float tmin = BAD_SCORE; int tpos = 0;
    const float* cv = cand_val + (size_t)(sneg * NQ + q) * NCAND;
    const int*   ci = cand_idx + (size_t)(sneg * NQ + q) * NCAND;
    for (int j = 0; j < NCAND; ++j)
      tk_insert<NB>(tv, ti, tmin, tpos, cv[j], ci[j]);

    // mask padded entities, map to global id, merge into final top-10
#pragma unroll
    for (int j = 0; j < NB; ++j) {
      const float s = tv[j] + (ti[j] >= SHARD_COUNT ? BAD_SCORE : 0.0f);
      const int idx = ti[j] < (MAX_ENT - 1) ? ti[j] : (MAX_ENT - 1);
      const int gid = table[(size_t)sneg * MAX_ENT + idx];
      tk_insert<KTOP>(fv, fg, fmin, fpos, s, gid);
    }
  }

  // sort descending (top_k returns sorted values)
  for (int a = 0; a < KTOP; ++a) {
    int best = a;
    for (int b = a + 1; b < KTOP; ++b)
      if (fv[b] > fv[best]) best = b;
    float tf = fv[a]; fv[a] = fv[best]; fv[best] = tf;
    int   tg = fg[a]; fg[a] = fg[best]; fg[best] = tg;
  }

#pragma unroll
  for (int k = 0; k < KTOP; ++k) {
    out_id[q * KTOP + k]  = fg[k];
    out_val[q * KTOP + k] = fv[k];
  }
}

// ---------------- launch ----------------
extern "C" void kernel_launch(void* const* d_in, const int* in_sizes, int n_in,
                              void* d_out, int out_size, void* d_ws, size_t ws_size,
                              hipStream_t stream) {
  (void)in_sizes; (void)n_in; (void)out_size; (void)ws_size;

  const int*   relation = (const int*)d_in[0];
  const int*   head     = (const int*)d_in[1];
  const float* ent      = (const float*)d_in[2];
  const float* rel      = (const float*)d_in[3];
  const int*   table    = (const int*)d_in[4];

  // workspace layout: hr bf16 (256 KB) | cand_val (5.5 MB) | cand_idx (5.5 MB)
  char* ws = (char*)d_ws;
  const size_t HR_BYTES = (size_t)NQ * EMB * sizeof(unsigned short);
  const size_t CAND_N   = (size_t)N_SHARD * NQ * NCAND;
  unsigned short* hr    = (unsigned short*)ws;
  float* cand_val       = (float*)(ws + HR_BYTES);
  int*   cand_idx       = (int*)(ws + HR_BYTES + CAND_N * sizeof(float));

  // outputs: ids (int32, 4*256*10) then values (f32, 4*256*10), flat-concatenated
  int*   out_id  = (int*)d_out;
  float* out_val = (float*)d_out + (size_t)N_SHARD * SHARD_BS * KTOP;

  hr_kernel<<<NQ, EMB, 0, stream>>>(relation, head, ent, rel, hr);

  dim3 grid2(NCHUNK, NQ / BLOCK_Q, N_SHARD);   // 16 x 4 x 4 = 256 WGs
  score_topk_kernel<<<grid2, 256, 0, stream>>>(hr, ent, cand_val, cand_idx);

  final_topk_kernel<<<NQ / 256, 256, 0, stream>>>(cand_val, cand_idx, table,
                                                  out_id, out_val);
}
